// Head_normal_74397423501932
// MI455X (gfx1250) — matrix operand order repaired and checked by reading, not compiled, not run on hardware
//
#include <hip/hip_runtime.h>
#include <hip/hip_fp16.h>

typedef __attribute__((ext_vector_type(16))) _Float16 v16h;
typedef __attribute__((ext_vector_type(8)))  _Float16 v8h;
typedef __attribute__((ext_vector_type(8)))  float    v8f;

#define SLEN   4096
#define NLAT   512
#define HEADD  42
#define HPAD   64
#define BATCH  8
#define BM     64     // queries per workgroup (4 waves x 16)
#define BN     64     // keys per LDS stage
#define KSTR   72     // LDS row stride in halfs (bank spread, keeps 16B align)
#define WSTR   520    // LDS stride for W^T rows (512 + 8)

__device__ __forceinline__ v8f wmma16(v16h a, v16h b, v8f c) {
  // v_wmma_f32_16x16x32_f16  D = A*B + C
  return __builtin_amdgcn_wmma_f32_16x16x32_f16(false, a, false, b, (short)0, c,
                                                false, false);
}

// Build a 16-half operand vector from two contiguous 8-half (16B) chunks.
__device__ __forceinline__ v16h ld2x8(const _Float16* p0, const _Float16* p1) {
  v8h lo = *(const v8h*)p0;
  v8h hi = *(const v8h*)p1;
  v16h r;
#pragma unroll
  for (int i = 0; i < 8; ++i) { r[i] = lo[i]; r[i + 8] = hi[i]; }
  return r;
}

// Async global->LDS 16-byte copy (GLOBAL_LOAD_ASYNC_TO_LDS_B128, ASYNCcnt).
__device__ __forceinline__ void async_copy_b128(unsigned lds_off, const void* g) {
  asm volatile("global_load_async_to_lds_b128 %0, %1, off"
               :
               : "v"(lds_off), "v"(g)
               : "memory");
}

__device__ __forceinline__ void wait_async0() {
#if __has_builtin(__builtin_amdgcn_s_wait_asynccnt)
  __builtin_amdgcn_s_wait_asynccnt(0);
#else
  asm volatile("s_wait_asynccnt 0x0" ::: "memory");
#endif
}

__device__ __forceinline__ unsigned lds_offset(const void* p) {
  return (unsigned)(unsigned long long)p;  // low 32 bits of generic LDS addr
}

// ---------------------------------------------------------------------------
// Projection: O[row, 0..63] = f16( (X @ W) * scale ), cols 42..63 zero padded.
// X: [32768, 512] f32, W: [512, 42] f32, O: [32768, 64] f16.
// One wave computes a 16-row tile; W^T (f16, pre-scaled) staged in LDS.
// ---------------------------------------------------------------------------
__global__ __launch_bounds__(256)
void proj_f16(const float* __restrict__ X, const float* __restrict__ W,
              _Float16* __restrict__ O, float scale) {
  __shared__ _Float16 WT[48 * WSTR];
  for (int idx = threadIdx.x; idx < 48 * NLAT; idx += 256) {
    int n = idx >> 9, k = idx & (NLAT - 1);
    float w = (n < HEADD) ? W[k * HEADD + n] * scale : 0.f;
    WT[n * WSTR + k] = (_Float16)w;
  }
  __syncthreads();

  const int wave = threadIdx.x >> 5, lane = threadIdx.x & 31;
  const int col = lane & 15, half = lane >> 4;
  const int row0 = blockIdx.x * 128 + wave * 16;
  const float* xr = X + (size_t)(row0 + col) * NLAT;

  v8f acc[3] = {};
  for (int kk = 0; kk < NLAT; kk += 32) {
    // A operand: element e -> k = kk + 16*(e/8) + half*8 + e%8
    v16h xa;
    const float* p0 = xr + kk + half * 8;
    const float* p1 = xr + kk + 16 + half * 8;
#pragma unroll
    for (int i = 0; i < 8; ++i) {
      xa[i]     = (_Float16)p0[i];
      xa[i + 8] = (_Float16)p1[i];
    }
#pragma unroll
    for (int t = 0; t < 3; ++t) {
      // B operand: lane col n = t*16+col; element e -> k = kk + 16*(e/8) + half*8 + e%8
      const _Float16* wr = &WT[(t * 16 + col) * WSTR + kk + half * 8];
      acc[t] = wmma16(xa, ld2x8(wr, wr + 16), acc[t]);
    }
  }
  // D: rows = x-rows (half*8+g), cols = t*16+col
#pragma unroll
  for (int t = 0; t < 3; ++t)
#pragma unroll
    for (int g = 0; g < 8; ++g)
      O[(size_t)(row0 + half * 8 + g) * HPAD + t * 16 + col] = (_Float16)acc[t][g];
#pragma unroll
  for (int g = 0; g < 8; ++g)
    O[(size_t)(row0 + half * 8 + g) * HPAD + 48 + col] = (_Float16)0.f;
}

// ---------------------------------------------------------------------------
// Flash attention. q/k/v: [B, S, 64] f16 (q pre-scaled by 42^-0.5, pad zeroed).
// out: [B, S, 42] f32. 128 threads = 4 waves; each wave owns 16 queries.
// Double-buffered LDS staging; K block copied with async global->LDS DMA.
// ---------------------------------------------------------------------------
__global__ __launch_bounds__(128)
void flash_attn(const _Float16* __restrict__ qp, const _Float16* __restrict__ kp,
                const _Float16* __restrict__ vp, float* __restrict__ out) {
  __shared__ _Float16 lK[2][BN * KSTR];     // K block, [key][h]
  __shared__ _Float16 lVT[2][HPAD * KSTR];  // V block transposed, [h][key]

  const int nqb  = SLEN / BM;
  const int b    = blockIdx.x / nqb;
  const int qblk = blockIdx.x % nqb;
  const int tid  = threadIdx.x;
  const int wave = tid >> 5, lane = tid & 31;
  const int col = lane & 15, half = lane >> 4;
  const size_t bbase = (size_t)b * SLEN * HPAD;
  const int qrow0 = qblk * BM + wave * 16;

  // Q as B-operand: lane col = query; element e -> h = kk*32 + 16*(e/8) + half*8 + e%8.
  const _Float16* qptr = qp + bbase + (size_t)(qrow0 + col) * HPAD + half * 8;
  const v16h qb0 = ld2x8(qptr, qptr + 16);
  const v16h qb1 = ld2x8(qptr + 32, qptr + 48);

  float m = -3.0e38f, l = 0.f;
  v8f o[3] = {};  // O tiles: rows = queries, cols = head (0..47 used)

  const int r = tid >> 1, part = tid & 1;  // cooperative loader mapping

  // Stage one 64-key block into buffer p: K via async DMA, V^T via scatter.
  auto stage = [&](int kb, int p) {
    const _Float16* gk = kp + bbase + (size_t)(kb + r) * HPAD + part * 32;
    const unsigned lk = lds_offset(&lK[p][r * KSTR + part * 32]);
#pragma unroll
    for (int i = 0; i < 4; ++i)
      async_copy_b128(lk + i * 16, (const void*)(gk + i * 8));

    const uint4* gv = (const uint4*)(vp + bbase + (size_t)(kb + r) * HPAD + part * 32);
    _Float16 vt[32];
#pragma unroll
    for (int i = 0; i < 4; ++i) ((uint4*)vt)[i] = gv[i];
#pragma unroll
    for (int i = 0; i < 32; ++i) lVT[p][(part * 32 + i) * KSTR + r] = vt[i];
  };

  stage(0, 0);

  for (int kb = 0; kb < SLEN; kb += BN) {
    const int p = (kb / BN) & 1;
    wait_async0();       // my async copies into buffer p are done
    __syncthreads();     // everyone's writes to p done; reads of p^1 retired

    if (kb + BN < SLEN) stage(kb + BN, p ^ 1);
    if (kb + 2 * BN < SLEN) {  // deep prefetch (global_prefetch_b8)
      __builtin_prefetch(vp + bbase + (size_t)(kb + 2 * BN) * HPAD + tid * 32, 0, 3);
    }

    const _Float16* Kb = lK[p];
    const _Float16* Vb = lVT[p];

#pragma unroll
    for (int c = 0; c < BN; c += 32) {
      // S^T = K * Q^T : two 16x16 score tiles (keys c..c+15, c+16..c+31)
      v8f s0 = {}, s1 = {};
      const _Float16* kr0 = &Kb[(c + col) * KSTR];
      const _Float16* kr1 = &Kb[(c + 16 + col) * KSTR];
      s0 = wmma16(ld2x8(kr0 + half * 8,      kr0 + 16 + half * 8), qb0, s0);
      s0 = wmma16(ld2x8(kr0 + 32 + half * 8, kr0 + 48 + half * 8), qb1, s0);
      s1 = wmma16(ld2x8(kr1 + half * 8,      kr1 + 16 + half * 8), qb0, s1);
      s1 = wmma16(ld2x8(kr1 + 32 + half * 8, kr1 + 48 + half * 8), qb1, s1);

      // Online softmax (per lane = one query column; halves exchange via xor16)
      float lmax = -3.0e38f;
#pragma unroll
      for (int g = 0; g < 8; ++g) lmax = fmaxf(lmax, fmaxf(s0[g], s1[g]));
      lmax = fmaxf(lmax, __shfl_xor(lmax, 16, 32));
      const float mnew  = fmaxf(m, lmax);
      const float alpha = exp2f((m - mnew) * 1.44269504f);

      v16h pa;  // P in A-operand layout: matches score D layout exactly
      float ls = 0.f;
#pragma unroll
      for (int g = 0; g < 8; ++g) {
        float p0 = exp2f((s0[g] - mnew) * 1.44269504f);
        float p1 = exp2f((s1[g] - mnew) * 1.44269504f);
        ls += p0 + p1;
        pa[g]     = (_Float16)p0;
        pa[g + 8] = (_Float16)p1;
      }
      ls += __shfl_xor(ls, 16, 32);
      l = l * alpha + ls;
      m = mnew;

      // Rescale O accumulators: O rows are queries half*8+g -> broadcast alpha
      float arow[8];
#pragma unroll
      for (int g = 0; g < 8; ++g) arow[g] = __shfl(alpha, half * 8 + g, 32);
#pragma unroll
      for (int t = 0; t < 3; ++t)
#pragma unroll
        for (int g = 0; g < 8; ++g) o[t][g] *= arow[g];

      // O += P * V : B operand from transposed V rows (contiguous in key)
#pragma unroll
      for (int t = 0; t < 3; ++t) {
        const _Float16* vr = &Vb[(t * 16 + col) * KSTR + c + half * 8];
        o[t] = wmma16(pa, ld2x8(vr, vr + 16), o[t]);
      }
    }
  }

  // Normalize and store: O rows = queries half*8+g, cols = head t*16+col
  float linv[8];
#pragma unroll
  for (int g = 0; g < 8; ++g) linv[g] = 1.0f / __shfl(l, half * 8 + g, 32);
  const size_t orow0 = (size_t)b * SLEN + qrow0;
#pragma unroll
  for (int t = 0; t < 3; ++t) {
    const int h = t * 16 + col;
    if (h < HEADD) {
#pragma unroll
      for (int g = 0; g < 8; ++g)
        out[(orow0 + half * 8 + g) * HEADD + h] = o[t][g] * linv[g];
    }
  }
}

// ---------------------------------------------------------------------------
extern "C" void kernel_launch(void* const* d_in, const int* in_sizes, int n_in,
                              void* d_out, int out_size, void* d_ws, size_t ws_size,
                              hipStream_t stream) {
  (void)in_sizes; (void)n_in; (void)out_size; (void)ws_size;
  const float* xk = (const float*)d_in[0];
  const float* xq = (const float*)d_in[1];
  const float* xv = (const float*)d_in[2];
  const float* Wk = (const float*)d_in[3];
  const float* Wq = (const float*)d_in[4];
  const float* Wv = (const float*)d_in[5];
  float* out = (float*)d_out;

  const size_t projElems = (size_t)BATCH * SLEN * HPAD;  // 2M halfs = 4MB each
  _Float16* qpf = (_Float16*)d_ws;
  _Float16* kpf = qpf + projElems;
  _Float16* vpf = kpf + projElems;

  const int rows = BATCH * SLEN;  // 32768
  dim3 pb(256), pg(rows / 128);
  const float scale = 0.15430334996209191f;  // 42^-0.5 folded into Q
  proj_f16<<<pg, pb, 0, stream>>>(xq, Wq, qpf, scale);
  proj_f16<<<pg, pb, 0, stream>>>(xk, Wk, kpf, 1.0f);
  proj_f16<<<pg, pb, 0, stream>>>(xv, Wv, vpf, 1.0f);

  dim3 ab(128), ag(BATCH * (SLEN / BM));  // 512 workgroups
  flash_attn<<<ag, ab, 0, stream>>>(qpf, kpf, vpf, out);
}
